// AttnBlock_16887811407979
// MI455X (gfx1250) — matrix operand order repaired and checked by reading, not compiled, not run on hardware
//
#include <hip/hip_runtime.h>
#include <hip/hip_bf16.h>

// ---- problem constants (match reference) ----
#define C_DIM   512
#define N_DIM   2048
#define KATT    128
#define HEADS   8
#define DHEAD   64
#define NGROUPS 32

// ---- vector types (plain ext vectors: trivially copyable, union-safe) ----
typedef __attribute__((ext_vector_type(16))) __bf16       v16bf;
typedef __attribute__((ext_vector_type(8)))  float        v8f;
typedef __attribute__((ext_vector_type(4)))  unsigned int u32x4;
typedef __attribute__((ext_vector_type(4)))  int          v4i;
typedef __attribute__((ext_vector_type(4)))  float        f32x4;
typedef __attribute__((ext_vector_type(2)))  float        f32x2;

// ---- gfx1250 async global->LDS path (ASYNCcnt), guarded by __has_builtin ----
#if defined(__AMDGCN__) && __has_builtin(__builtin_amdgcn_global_load_async_to_lds_b128) && \
    __has_builtin(__builtin_amdgcn_s_wait_asynccnt)
#define HAVE_ASYNC_LDS 1
#else
#define HAVE_ASYNC_LDS 0
#endif

// builtin signature (from hipcc diagnostic): params are v4i pointers with
// explicit address spaces: src = AS(1) global, dst = AS(3) LDS.
typedef __attribute__((address_space(1))) v4i v4i_glb;
typedef __attribute__((address_space(3))) v4i v4i_lds;

// flat->LDS addrspacecast lowers to low-32-bit truncation (ISA aperture rule),
// so build the AS(3)/AS(1) pointers via integer round-trip.
__device__ __forceinline__ v4i_glb* to_glb_v4i(const void* p) {
  return (v4i_glb*)(unsigned long long)p;
}
__device__ __forceinline__ v4i_lds* to_lds_v4i(void* p) {
  return (v4i_lds*)(unsigned int)(unsigned long long)p;
}

__device__ __forceinline__ unsigned short f32_to_bf16(float f) {
  unsigned int u = __float_as_uint(f);
  u += 0x7FFFu + ((u >> 16) & 1u);           // round-to-nearest-even
  return (unsigned short)(u >> 16);
}

// =====================================================================
// 1) GroupNorm:  x (C,N) f32  ->  hT (N,C) bf16   (transposed for WMMA B)
// =====================================================================
__global__ void gn_kernel(const float* __restrict__ x,
                          const float* __restrict__ w,
                          const float* __restrict__ b,
                          unsigned short* __restrict__ hT) {
  const int g   = blockIdx.x;
  const int tid = threadIdx.x;
  const int c0  = g * 16;
  __shared__ float s_sum[256], s_sq[256];

  float sum = 0.f, sq = 0.f;
  for (int i = tid; i < 16 * N_DIM; i += 256) {
    int c = i >> 11, n = i & (N_DIM - 1);
    float v = x[(c0 + c) * N_DIM + n];
    sum += v; sq += v * v;
  }
  s_sum[tid] = sum; s_sq[tid] = sq;
  __syncthreads();
  for (int off = 128; off > 0; off >>= 1) {
    if (tid < off) { s_sum[tid] += s_sum[tid + off]; s_sq[tid] += s_sq[tid + off]; }
    __syncthreads();
  }
  const float inv  = 1.0f / (16.0f * (float)N_DIM);
  const float mean = s_sum[0] * inv;
  const float var  = s_sq[0] * inv - mean * mean;
  const float rstd = rsqrtf(var + 1e-6f);

  for (int i = tid; i < 16 * N_DIM; i += 256) {
    int c = i >> 11, n = i & (N_DIM - 1);
    int cc = c0 + c;
    float v = x[cc * N_DIM + n];
    float y = (v - mean) * rstd * w[cc] + b[cc];
    hT[n * C_DIM + cc] = f32_to_bf16(y);
  }
}

// =====================================================================
// 2) f32 -> bf16 weight conversion (row-major preserved)
// =====================================================================
__global__ void cvt_bf16_kernel(const float* __restrict__ src,
                                unsigned short* __restrict__ dst, int n) {
  int i = blockIdx.x * blockDim.x + threadIdx.x;
  if (i < n) dst[i] = f32_to_bf16(src[i]);
}

// =====================================================================
// 3) WMMA GEMM core.  WG tile 128(M) x 64(N); 8 waves as 4(M) x 2(N);
//    wave tile 32x32 (2x2 register blocking -> 4 v_wmma per K-step).
//    The 64-row B panel (full K=512, bf16) is staged once per WG into
//    64KB of LDS via GLOBAL_LOAD_ASYNC_TO_LDS_B128 (ASYNCcnt), with
//    16B-chunk XOR swizzle (chunk ^ row&15) for conflict-free
//    ds_load_b128 fragment reads.  A fragments come straight from L2.
// =====================================================================
union FragBF { v16bf v; u32x4 u[2]; };

#define WMMA_BF16(Afrag, Bfrag, Cacc) \
  __builtin_amdgcn_wmma_f32_16x16x32_bf16(false, (Afrag).v, false, (Bfrag).v, \
                                          (short)0, (Cacc), false, false)

struct Acc4 { v8f a00, a01, a10, a11; };

__device__ __forceinline__ void gemm_block_compute(const unsigned short* __restrict__ A,
                                                   const unsigned short* __restrict__ Bt,
                                                   int mblk_base, int nblk_base,
                                                   Acc4& r, int& m0_out, int& n0_out) {
  __shared__ unsigned short sB[64 * C_DIM];          // 64KB B panel
  const int tid  = threadIdx.x;
  const int wave = tid >> 5;
  const int lane = tid & 31;
  const int half = lane >> 4;
  const int l16  = lane & 15;
  const int m0 = mblk_base + (wave & 3) * 32;
  const int n0 = nblk_base + (wave >> 2) * 32;
  m0_out = m0; n0_out = n0;

  // ---- stage B panel: 64 rows x 512 K, 16B chunks, row-XOR swizzled ----
#pragma unroll
  for (int i = 0; i < 16; ++i) {
    const int ch = tid + 256 * i;                    // 4096 chunks total
    const int rr = ch >> 6;                          // row 0..63
    const int cc = ch & 63;                          // chunk 0..63 in row
    const unsigned short* gp = Bt + (size_t)(nblk_base + rr) * C_DIM + cc * 8;
    unsigned short*       lp = sB + rr * C_DIM + ((cc ^ (rr & 15)) << 3);
#if HAVE_ASYNC_LDS
    __builtin_amdgcn_global_load_async_to_lds_b128(to_glb_v4i(gp), to_lds_v4i(lp), 0, 0);
#else
    *reinterpret_cast<u32x4*>(lp) = *reinterpret_cast<const u32x4*>(gp);
#endif
  }
#if HAVE_ASYNC_LDS
  __builtin_amdgcn_s_wait_asynccnt(0);
#endif
  __syncthreads();

  const unsigned short* a0p = A + (size_t)(m0 + l16) * C_DIM;
  const unsigned short* a1p = a0p + 16 * C_DIM;
  const int bl0 = (n0 - nblk_base) + l16;            // local B row 0..63
  const unsigned short* sb0 = sB + bl0 * C_DIM;
  const unsigned short* sb1 = sb0 + 16 * C_DIM;
  const int swz = l16;                               // == row&15 for both rows

  __builtin_prefetch(a0p, 0, 3);                     // global_prefetch_b8
  __builtin_prefetch(a1p, 0, 3);

#pragma unroll 2
  for (int kk = 0; kk < C_DIM; kk += 32) {
    const int c0 = (kk >> 3) + half;                 // logical chunk
    FragBF fa0, fa1, fb0, fb1;
    // A-frag (ISA 7.12.2): lane m=l16; K = kk+8*half+[0..7], then +16
    fa0.u[0] = *reinterpret_cast<const u32x4*>(a0p + kk + 8 * half);
    fa0.u[1] = *reinterpret_cast<const u32x4*>(a0p + kk + 16 + 8 * half);
    fa1.u[0] = *reinterpret_cast<const u32x4*>(a1p + kk + 8 * half);
    fa1.u[1] = *reinterpret_cast<const u32x4*>(a1p + kk + 16 + 8 * half);
    // B-frag from LDS (swizzled chunks): K = kk+8*half+[0..7], then +16
    fb0.u[0] = *reinterpret_cast<const u32x4*>(sb0 + (((c0    ) ^ swz) << 3));
    fb0.u[1] = *reinterpret_cast<const u32x4*>(sb0 + (((c0 + 2) ^ swz) << 3));
    fb1.u[0] = *reinterpret_cast<const u32x4*>(sb1 + (((c0    ) ^ swz) << 3));
    fb1.u[1] = *reinterpret_cast<const u32x4*>(sb1 + (((c0 + 2) ^ swz) << 3));

    r.a00 = WMMA_BF16(fa0, fb0, r.a00);
    r.a01 = WMMA_BF16(fa0, fb1, r.a01);
    r.a10 = WMMA_BF16(fa1, fb0, r.a10);
    r.a11 = WMMA_BF16(fa1, fb1, r.a11);
  }
}

// D layout: lane column n=l16; VGPR r -> row m = mbase + r, mbase = +8*half
__device__ __forceinline__ void store_tile_T(float* __restrict__ outT,
                                             const float* __restrict__ bias,
                                             const v8f& acc, int mbase, int ncol) {
  f32x4 lo = { acc[0] + bias[mbase + 0], acc[1] + bias[mbase + 1],
               acc[2] + bias[mbase + 2], acc[3] + bias[mbase + 3] };
  f32x4 hi = { acc[4] + bias[mbase + 4], acc[5] + bias[mbase + 5],
               acc[6] + bias[mbase + 6], acc[7] + bias[mbase + 7] };
  *reinterpret_cast<f32x4*>(outT + (size_t)ncol * C_DIM + mbase)     = lo;
  *reinterpret_cast<f32x4*>(outT + (size_t)ncol * C_DIM + mbase + 4) = hi;
}

// ---------------------------------------------------------------------
// 3a) fused q/k/v GEMM: blockIdx.y selects {wq,wk,wv}; outputs are the
//     contiguous qT/kT/vT f32 [n][c] buffers (gather-friendly layout).
// ---------------------------------------------------------------------
__global__ void gemm_qkv_kernel(const unsigned short* __restrict__ Wb3,
                                const unsigned short* __restrict__ hT,
                                const float* __restrict__ bq,
                                const float* __restrict__ bk,
                                const float* __restrict__ bv,
                                float* __restrict__ outT3) {
  const int which = blockIdx.y;
  const unsigned short* A = Wb3 + (size_t)which * C_DIM * C_DIM;
  const float* bias = (which == 0) ? bq : (which == 1) ? bk : bv;
  float* outT = outT3 + (size_t)which * N_DIM * C_DIM;

  const int lane = threadIdx.x & 31;
  const int half = lane >> 4, l16 = lane & 15;
  Acc4 r; r.a00 = {}; r.a01 = {}; r.a10 = {}; r.a11 = {};
  int m0, n0;
  gemm_block_compute(A, hT, (blockIdx.x & 3) * 128, (blockIdx.x >> 2) * 64, r, m0, n0);

  store_tile_T(outT, bias, r.a00, m0 +      8 * half, n0 +      l16);
  store_tile_T(outT, bias, r.a01, m0 +      8 * half, n0 + 16 + l16);
  store_tile_T(outT, bias, r.a10, m0 + 16 + 8 * half, n0 +      l16);
  store_tile_T(outT, bias, r.a11, m0 + 16 + 8 * half, n0 + 16 + l16);
}

// ---------------------------------------------------------------------
// 3b) projection GEMM + bias + residual -> d_out (C,N) f32
// ---------------------------------------------------------------------
__global__ void gemm_proj_kernel(const unsigned short* __restrict__ Wpb,
                                 const unsigned short* __restrict__ oT,
                                 const float* __restrict__ bias,
                                 const float* __restrict__ x,
                                 float* __restrict__ out) {
  const int lane = threadIdx.x & 31;
  const int half = lane >> 4, l16 = lane & 15;
  Acc4 r; r.a00 = {}; r.a01 = {}; r.a10 = {}; r.a11 = {};
  int m0, n0;
  gemm_block_compute(Wpb, oT, (blockIdx.x & 3) * 128, (blockIdx.x >> 2) * 64, r, m0, n0);

  const v8f* accs[4] = { &r.a00, &r.a01, &r.a10, &r.a11 };
#pragma unroll
  for (int t = 0; t < 4; ++t) {
    const int mbase = m0 + ((t >> 1) ? 16 : 0) + 8 * half;
    const int ncol  = n0 + ((t & 1) ? 16 : 0) + l16;
    const v8f& a = *accs[t];
#pragma unroll
    for (int rr = 0; rr < 8; ++rr) {
      const int m = mbase + rr;
      out[(size_t)m * N_DIM + ncol] = x[(size_t)m * N_DIM + ncol] + a[rr] + bias[m];
    }
  }
}

// =====================================================================
// 4) Sparse gathered attention. One WG per query n, one wave per head.
// =====================================================================
__global__ void attn_kernel(const float* __restrict__ qT,
                            const float* __restrict__ kT,
                            const float* __restrict__ vT,
                            const int* __restrict__ vmask,
                            const int* __restrict__ aidx,
                            unsigned short* __restrict__ oT) {
  const int n    = blockIdx.x;
  const int h    = threadIdx.x >> 5;
  const int lane = threadIdx.x & 31;

  __shared__ float sq[HEADS][DHEAD];
  __shared__ float sw[HEADS][KATT];
  __shared__ int   sidx[KATT];

  const float* qp = qT + n * C_DIM + h * DHEAD;
  sq[h][2 * lane]     = qp[2 * lane];
  sq[h][2 * lane + 1] = qp[2 * lane + 1];
  if (h == 0) {
#pragma unroll
    for (int j = 0; j < 4; ++j)
      sidx[lane + 32 * j] = aidx[n * KATT + lane + 32 * j];
  }
  __syncthreads();

  // ---- phase 1: scores + softmax over K=128 ----
  float sc[4];
#pragma unroll
  for (int j = 0; j < 4; ++j) {
    const int kk  = lane + 32 * j;
    const int idx = sidx[kk];
    const int msk = vmask[n * KATT + kk];
    const float* kp = kT + idx * C_DIM + h * DHEAD;
    float s = 0.f;
#pragma unroll
    for (int d = 0; d < DHEAD; d += 4) {
      f32x4 kv = *reinterpret_cast<const f32x4*>(kp + d);
      s += sq[h][d] * kv.x + sq[h][d + 1] * kv.y +
           sq[h][d + 2] * kv.z + sq[h][d + 3] * kv.w;
    }
    sc[j] = msk ? s : -__builtin_inff();
  }
  float mx = fmaxf(fmaxf(sc[0], sc[1]), fmaxf(sc[2], sc[3]));
  for (int off = 16; off > 0; off >>= 1) mx = fmaxf(mx, __shfl_xor(mx, off, 32));
  float e[4], ssum = 0.f;
#pragma unroll
  for (int j = 0; j < 4; ++j) { e[j] = __expf(sc[j] - mx); ssum += e[j]; }
  for (int off = 16; off > 0; off >>= 1) ssum += __shfl_xor(ssum, off, 32);
  const float rinv = 1.0f / ssum;
#pragma unroll
  for (int j = 0; j < 4; ++j) sw[h][lane + 32 * j] = e[j] * rinv;
  __syncthreads();

  // ---- phase 2: o[d] = sum_k w[k] * v[idx[k]][h*64+d]  (coalesced) ----
  const int d0 = lane * 2;
  float a0 = 0.f, a1 = 0.f;
  for (int k = 0; k < KATT; ++k) {
    const float wgt = sw[h][k];
    const float* vp = vT + sidx[k] * C_DIM + h * DHEAD;
    f32x2 vv = *reinterpret_cast<const f32x2*>(vp + d0);
    a0 += wgt * vv.x;
    a1 += wgt * vv.y;
  }
  // reference: o transposed (b,d,h,q).reshape -> channel c = d*HEADS + h
  oT[n * C_DIM + (d0 * HEADS + h)]       = f32_to_bf16(a0);
  oT[n * C_DIM + ((d0 + 1) * HEADS + h)] = f32_to_bf16(a1);
}

// =====================================================================
// launcher
// =====================================================================
extern "C" void kernel_launch(void* const* d_in, const int* in_sizes, int n_in,
                              void* d_out, int out_size, void* d_ws, size_t ws_size,
                              hipStream_t stream) {
  const float* x     = (const float*)d_in[0];
  const int*   vmask = (const int*)  d_in[1];
  const int*   aidx  = (const int*)  d_in[2];
  const float* gnw   = (const float*)d_in[3];
  const float* gnb   = (const float*)d_in[4];
  const float* wq    = (const float*)d_in[5];
  const float* bq    = (const float*)d_in[6];
  const float* wk    = (const float*)d_in[7];
  const float* bk    = (const float*)d_in[8];
  const float* wv    = (const float*)d_in[9];
  const float* bv    = (const float*)d_in[10];
  const float* wp    = (const float*)d_in[11];
  const float* bp    = (const float*)d_in[12];
  float* out = (float*)d_out;

  // workspace carve-up (~18 MB); wq/wk/wv bf16 and qT/kT/vT kept contiguous
  char* ws = (char*)d_ws;
  unsigned short* hT  = (unsigned short*)ws; ws += (size_t)N_DIM * C_DIM * 2;
  unsigned short* wb3 = (unsigned short*)ws; ws += 3 * (size_t)C_DIM * C_DIM * 2;
  unsigned short* wpb = (unsigned short*)ws; ws += (size_t)C_DIM * C_DIM * 2;
  float* qkvT = (float*)ws;                  ws += 3 * (size_t)N_DIM * C_DIM * 4;
  unsigned short* oT = (unsigned short*)ws;  ws += (size_t)N_DIM * C_DIM * 2;

  float* qT = qkvT;
  float* kT = qkvT + (size_t)N_DIM * C_DIM;
  float* vT = qkvT + 2 * (size_t)N_DIM * C_DIM;

  // 1) GroupNorm -> hT (bf16, transposed)
  gn_kernel<<<NGROUPS, 256, 0, stream>>>(x, gnw, gnb, hT);

  // 2) weights -> bf16 (wq,wk,wv into contiguous block, wp separate)
  const int wElems = C_DIM * C_DIM;
  cvt_bf16_kernel<<<(wElems + 255) / 256, 256, 0, stream>>>(wq, wb3,              wElems);
  cvt_bf16_kernel<<<(wElems + 255) / 256, 256, 0, stream>>>(wk, wb3 + wElems,     wElems);
  cvt_bf16_kernel<<<(wElems + 255) / 256, 256, 0, stream>>>(wv, wb3 + 2 * wElems, wElems);
  cvt_bf16_kernel<<<(wElems + 255) / 256, 256, 0, stream>>>(wp, wpb,              wElems);

  // 3) fused q/k/v WMMA GEMM: 128 WGs x 3, WG tile 128x64 with async-LDS
  //    B-panel staging, wave tile 32x32 (4 v_wmma per K-step)
  gemm_qkv_kernel<<<dim3(128, 3), 256, 0, stream>>>(wb3, hT, bq, bk, bv, qkvT);

  // 4) sparse attention (one WG per query, one wave per head)
  attn_kernel<<<N_DIM, 256, 0, stream>>>(qT, kT, vT, vmask, aidx, oT);

  // 5) projection GEMM + residual -> d_out (C,N) f32
  gemm_proj_kernel<<<128, 256, 0, stream>>>(wpb, oT, bp, x, out);
}
